// CenterBBoxHead_42210938585428
// MI455X (gfx1250) — compile-verified, matrix operand order checked by reading, not run on hardware
//
#include <hip/hip_runtime.h>
#include <hip/hip_bf16.h>

typedef __bf16 bf16_t;
typedef __attribute__((ext_vector_type(16))) __bf16 v16bf;
typedef __attribute__((ext_vector_type(8)))  __bf16 v8bf;
typedef __attribute__((ext_vector_type(8)))  float  v8f;
typedef int b128_t __attribute__((vector_size(16)));   // matches async builtin param

constexpr int NB   = 4;
constexpr int C_IN = 384;
constexpr int C_SH = 64;
constexpr int HT   = 256;
constexpr int WT   = 256;
constexpr int HWSZ = HT * WT;
constexpr int NCLS = 3;
constexpr int KTOP = 500;

// CDNA5 async global->LDS path if the toolchain exposes the builtins.
#if defined(__has_builtin)
#  if __has_builtin(__builtin_amdgcn_global_load_async_to_lds_b128) && \
      __has_builtin(__builtin_amdgcn_s_wait_asynccnt)
#    define USE_ASYNC_LDS 1
#  endif
#endif

// ---------------------------------------------------------------------------
// Reorder + convert conv weights: [co][ci][ky][kx] fp32 ->
//   [cc][tap][co][ci32] bf16 (chunk-major, contiguous 36KB per cin-chunk)
// so the per-chunk LDS weight stage is a straight b128 copy.
// ---------------------------------------------------------------------------
__global__ __launch_bounds__(256) void reorder_w_kernel(
    const float* __restrict__ in, bf16_t* __restrict__ out, int cint) {
  int i = blockIdx.x * 256 + threadIdx.x;
  int total = 64 * cint * 9;
  if (i >= total) return;
  int ci32  = i & 31;
  int co    = (i >> 5) & 63;
  int rest  = i >> 11;        // cc*9 + tap
  int tap   = rest % 9;
  int chunk = rest / 9;
  out[i] = (bf16_t)in[(long)(co * cint + chunk * 32 + ci32) * 9 + tap];
}

// ---------------------------------------------------------------------------
// Implicit-GEMM 3x3 conv (SAME) + BN + ReLU via v_wmma_f32_16x16x32_bf16.
// Block = 64 couts x 2 rows x 32 pixels. 8 waves = 4 cout groups x 2 rows;
// each wave holds two 16x16 accumulators (pixel tiles 0-15 / 16-31).
// LDS layouts keep every fragment as two contiguous 16-byte runs:
//   sA [tap][co][ci]   weights (async b128 staged when available)
//   sBt[row][col][ci]  input halo (4 rows for the 2 output rows)
// Per tap: A loaded once, reused for both B tiles -> 3 b128 loads / WMMA.
// ---------------------------------------------------------------------------
template <int CINT, bool IN_F32>
__global__ __launch_bounds__(256) void conv3x3_bn_relu_wmma(
    const float*  __restrict__ xf,    // fp32 input (IN_F32)
    const bf16_t* __restrict__ xb,    // bf16 input (!IN_F32)
    const bf16_t* __restrict__ wtr,   // reordered [cc][tap][64][32] bf16
    const float* __restrict__ gamma, const float* __restrict__ beta,
    const float* __restrict__ mean,  const float* __restrict__ var,
    bf16_t* __restrict__ out)         // [NB][64][HT][WT] bf16
{
  __shared__ __align__(16) bf16_t sA[9][64][32];    // 36864 B
  __shared__ __align__(16) bf16_t sBt[4][36][32];   //  9216 B

  const int tid  = threadIdx.x;
  const int wave = tid >> 5;
  const int lane = tid & 31;
  const int cg   = wave & 3;          // cout group
  const int rr   = wave >> 2;         // output row within block (0/1)

  const int tiles = WT / 32;
  int bid = blockIdx.x;
  int txt = bid % tiles;
  int y0  = ((bid / tiles) % (HT / 2)) * 2;
  int bb  = bid / (tiles * (HT / 2));
  int x0  = txt * 32;

  v8f acc0 = {0.f, 0.f, 0.f, 0.f, 0.f, 0.f, 0.f, 0.f};
  v8f acc1 = {0.f, 0.f, 0.f, 0.f, 0.f, 0.f, 0.f, 0.f};

  const int m     = lane & 15;
  const int khalf = (lane & 16) ? 8 : 0;
  const int arow  = cg * 16 + m;

  for (int cc = 0; cc < CINT / 32; ++cc) {
    // ---- stage input halo: 32 ci x 4 rows x 34 cols (boundary zero-fill) ----
    for (int e = tid; e < 32 * 4 * 34; e += 256) {
      int ci  = e / 136;
      int rem = e - ci * 136;
      int r   = rem / 34;
      int c   = rem - r * 34;
      int gy = y0 + r - 1, gx = x0 + c - 1;
      float v = 0.f;
      if (gy >= 0 && gy < HT && gx >= 0 && gx < WT) {
        long idx = ((long)bb * CINT + cc * 32 + ci) * HWSZ + (long)gy * WT + gx;
        v = IN_F32 ? xf[idx] : (float)xb[idx];
      }
      sBt[r][c][ci] = (bf16_t)v;
    }

    // ---- stage weight slice: contiguous 36KB block for this chunk ----
    const bf16_t* wsrc = wtr + (size_t)cc * (9 * 64 * 32);
#ifdef USE_ASYNC_LDS
    {
      b128_t __attribute__((address_space(1)))* g =
          (b128_t __attribute__((address_space(1)))*)(void*)wsrc;
      b128_t __attribute__((address_space(3)))* l =
          (b128_t __attribute__((address_space(3)))*)(void*)&sA[0][0][0];
      for (int e = tid; e < (9 * 64 * 32) / 8; e += 256)
        __builtin_amdgcn_global_load_async_to_lds_b128(g + e, l + e, 0, 0);
      __builtin_amdgcn_s_wait_asynccnt(0);
    }
#else
    for (int e = tid; e < (9 * 64 * 32) / 8; e += 256)
      ((v8bf*)&sA[0][0][0])[e] = ((const v8bf*)wsrc)[e];
#endif

    // prefetch next chunk's input rows into cache
    if (cc + 1 < CINT / 32) {
      long pidx = ((long)bb * CINT + (cc + 1) * 32) * HWSZ + (long)y0 * WT + x0;
      __builtin_prefetch(IN_F32 ? (const void*)(xf + pidx)
                                : (const void*)(xb + pidx), 0, 0);
    }
    __syncthreads();

    #pragma unroll
    for (int tap = 0; tap < 9; ++tap) {
      int ky = tap / 3, kx = tap - ky * 3;
      // A fragment: shared by both pixel tiles of this wave
      v8bf alo = *(const v8bf*)&sA[tap][arow][khalf];
      v8bf ahi = *(const v8bf*)&sA[tap][arow][16 + khalf];
      v16bf a = __builtin_shufflevector(alo, ahi, 0, 1, 2, 3, 4, 5, 6, 7,
                                        8, 9, 10, 11, 12, 13, 14, 15);
      int r = rr + ky;           // halo row for this wave's output row
      // B fragment, pixel tile 0 (cols 0-15)
      int c0 = m + kx;
      v8bf b0lo = *(const v8bf*)&sBt[r][c0][khalf];
      v8bf b0hi = *(const v8bf*)&sBt[r][c0][16 + khalf];
      v16bf b0 = __builtin_shufflevector(b0lo, b0hi, 0, 1, 2, 3, 4, 5, 6, 7,
                                         8, 9, 10, 11, 12, 13, 14, 15);
      acc0 = __builtin_amdgcn_wmma_f32_16x16x32_bf16(
          false, a, false, b0, (short)0, acc0, false, false);
      // B fragment, pixel tile 1 (cols 16-31)
      int c1 = 16 + m + kx;
      v8bf b1lo = *(const v8bf*)&sBt[r][c1][khalf];
      v8bf b1hi = *(const v8bf*)&sBt[r][c1][16 + khalf];
      v16bf b1 = __builtin_shufflevector(b1lo, b1hi, 0, 1, 2, 3, 4, 5, 6, 7,
                                         8, 9, 10, 11, 12, 13, 14, 15);
      acc1 = __builtin_amdgcn_wmma_f32_16x16x32_bf16(
          false, a, false, b1, (short)0, acc1, false, false);
    }
    __syncthreads();
  }

  // ---- epilogue: BN + ReLU, store bf16 (2 tiles per wave) ----
  int n    = lane & 15;
  int yrow = y0 + rr;
  #pragma unroll
  for (int j = 0; j < 8; ++j) {
    int co = cg * 16 + j + ((lane & 16) ? 8 : 0);
    float s = gamma[co] * rsqrtf(var[co] + 1e-5f);
    float v0 = fmaxf((acc0[j] - mean[co]) * s + beta[co], 0.f);
    float v1 = fmaxf((acc1[j] - mean[co]) * s + beta[co], 0.f);
    long base = ((long)bb * C_SH + co) * HWSZ + (long)yrow * WT + x0;
    out[base + n]      = (bf16_t)v0;
    out[base + 16 + n] = (bf16_t)v1;
  }
}

// ---------------------------------------------------------------------------
// Tiny task conv: 3x3, 64 -> cout (1..3), + bias, fp32 straight to d_out.
// ---------------------------------------------------------------------------
__global__ __launch_bounds__(256) void conv3x3_head_kernel(
    const bf16_t* __restrict__ hin, const float* __restrict__ w2,
    const float* __restrict__ b2, float* __restrict__ outp, int cout)
{
  __shared__ float sw[3 * 64 * 9];
  __shared__ float sb2[3];
  int tid = threadIdx.x;
  for (int e = tid; e < cout * 64 * 9; e += 256) sw[e] = w2[e];
  if (tid < cout) sb2[tid] = b2[tid];
  __syncthreads();

  long p = (long)blockIdx.x * 256 + tid;
  if (p >= (long)NB * HWSZ) return;
  int bb  = (int)(p / HWSZ);
  int rem = (int)(p % HWSZ);
  int y = rem / WT, x = rem % WT;

  float acc[3];
  for (int c = 0; c < 3; ++c) acc[c] = (c < cout) ? sb2[c] : 0.f;

  for (int ci = 0; ci < 64; ++ci) {
    const bf16_t* hp = hin + ((long)bb * 64 + ci) * HWSZ;
    #pragma unroll
    for (int ky = 0; ky < 3; ++ky) {
      int gy = y + ky - 1;
      if (gy < 0 || gy >= HT) continue;
      #pragma unroll
      for (int kx = 0; kx < 3; ++kx) {
        int gx = x + kx - 1;
        if (gx < 0 || gx >= WT) continue;
        float v = (float)hp[gy * WT + gx];
        int t = ky * 3 + kx;
        for (int c = 0; c < cout; ++c) acc[c] += v * sw[(c * 64 + ci) * 9 + t];
      }
    }
  }
  for (int c = 0; c < cout; ++c)
    outp[((long)bb * cout + c) * HWSZ + rem] = acc[c];
}

// ---------------------------------------------------------------------------
__global__ __launch_bounds__(256) void sigmoid_kernel(
    const float* __restrict__ in, float* __restrict__ out, int n) {
  int i = blockIdx.x * 256 + threadIdx.x;
  if (i < n) out[i] = 1.f / (1.f + __expf(-in[i]));
}

// Per-(b,class) top-500 by iterative argmax with suppression bitmap.
__global__ __launch_bounds__(256) void topk_class_kernel(
    const float* __restrict__ scores, int* __restrict__ taken,
    float* __restrict__ out_s, int* __restrict__ out_i)
{
  int row = blockIdx.x;  // b*NCLS + c
  const float* s = scores + (long)row * HWSZ;
  int* tk = taken + (long)row * HWSZ;
  int tid = threadIdx.x;
  for (int i = tid; i < HWSZ; i += 256) tk[i] = 0;
  __syncthreads();
  __shared__ float ms[256];
  __shared__ int   mi[256];
  for (int k = 0; k < KTOP; ++k) {
    float best = -1e30f; int bi = 0;
    for (int i = tid; i < HWSZ; i += 256) {
      float v = s[i];
      if (!tk[i] && v > best) { best = v; bi = i; }
    }
    ms[tid] = best; mi[tid] = bi;
    __syncthreads();
    for (int off = 128; off > 0; off >>= 1) {
      if (tid < off && ms[tid + off] > ms[tid]) {
        ms[tid] = ms[tid + off]; mi[tid] = mi[tid + off];
      }
      __syncthreads();
    }
    if (tid == 0) {
      out_s[(long)row * KTOP + k] = ms[0];
      out_i[(long)row * KTOP + k] = mi[0];
      tk[mi[0]] = 1;
    }
    __syncthreads();
  }
}

// Per-batch top-500 across classes (1500 candidates).
__global__ __launch_bounds__(256) void topk_merge_kernel(
    const float* __restrict__ cls_s, const int* __restrict__ cls_i,
    float* __restrict__ fsc, int* __restrict__ find, int* __restrict__ fcls)
{
  const int NTOT = NCLS * KTOP;
  int bb = blockIdx.x, tid = threadIdx.x;
  __shared__ float sc[NCLS * KTOP];
  __shared__ unsigned char used[NCLS * KTOP];
  __shared__ float ms[256];
  __shared__ int   mi[256];
  for (int i = tid; i < NTOT; i += 256) {
    sc[i] = cls_s[(long)bb * NTOT + i];
    used[i] = 0;
  }
  __syncthreads();
  for (int k = 0; k < KTOP; ++k) {
    float best = -1e30f; int bi = 0;
    for (int i = tid; i < NTOT; i += 256)
      if (!used[i] && sc[i] > best) { best = sc[i]; bi = i; }
    ms[tid] = best; mi[tid] = bi;
    __syncthreads();
    for (int off = 128; off > 0; off >>= 1) {
      if (tid < off && ms[tid + off] > ms[tid]) {
        ms[tid] = ms[tid + off]; mi[tid] = mi[tid + off];
      }
      __syncthreads();
    }
    if (tid == 0) {
      int j = mi[0];
      used[j] = 1;
      fsc[(long)bb * KTOP + k]  = ms[0];
      fcls[(long)bb * KTOP + k] = j / KTOP;
      find[(long)bb * KTOP + k] = cls_i[(long)bb * NTOT + j];
    }
    __syncthreads();
  }
}

// Gather regression maps at top-K indices, build boxes + validity.
__global__ __launch_bounds__(512) void decode_kernel(
    const float* __restrict__ outbuf, const float* __restrict__ fsc,
    const int* __restrict__ find, const int* __restrict__ fcls,
    float* __restrict__ boxes, int* __restrict__ keep)
{
  const long OFF_CENTER = 0;
  const long OFF_CZ  = (long)NB * 2 * HWSZ;
  const long OFF_DIM = OFF_CZ + (long)NB * 1 * HWSZ;
  const long OFF_ROT = OFF_DIM + (long)NB * 3 * HWSZ;
  int bb = blockIdx.x;
  int k  = threadIdx.x;
  if (k >= KTOP) return;
  long kk = (long)bb * KTOP + k;
  int ind = find[kk];
  float score = fsc[kk];
  float ys = (float)(ind / WT), xs = (float)(ind % WT);
  float cx = outbuf[OFF_CENTER + ((long)bb * 2 + 0) * HWSZ + ind];
  float cy = outbuf[OFF_CENTER + ((long)bb * 2 + 1) * HWSZ + ind];
  float z  = outbuf[OFF_CZ + (long)bb * HWSZ + ind];
  float d0 = __expf(outbuf[OFF_DIM + ((long)bb * 3 + 0) * HWSZ + ind]);
  float d1 = __expf(outbuf[OFF_DIM + ((long)bb * 3 + 1) * HWSZ + ind]);
  float d2 = __expf(outbuf[OFF_DIM + ((long)bb * 3 + 2) * HWSZ + ind]);
  float rc = outbuf[OFF_ROT + ((long)bb * 2 + 0) * HWSZ + ind];
  float rs = outbuf[OFF_ROT + ((long)bb * 2 + 1) * HWSZ + ind];
  float ang = atan2f(rs, rc);
  float xw = (xs + cx) * (2.f * 0.32f) + (-75.2f);
  float yw = (ys + cy) * (2.f * 0.32f) + (-75.2f);
  float* bx = boxes + kk * 7;
  bx[0] = xw; bx[1] = yw; bx[2] = z;
  bx[3] = d0; bx[4] = d1; bx[5] = d2; bx[6] = ang;
  bool inr = xw >= -80.f && xw <= 80.f && yw >= -80.f && yw <= 80.f &&
             z >= -10.f && z <= 10.f;
  keep[kk] = (score > 0.1f && inr) ? 1 : 0;
  (void)fcls;
}

// Greedy circle-NMS + emit padded rois/scores/labels.
__global__ __launch_bounds__(512) void nms_emit_kernel(
    const float* __restrict__ boxes, const float* __restrict__ fsc,
    const int* __restrict__ fcls, const int* __restrict__ keep,
    float* __restrict__ rois, float* __restrict__ roi_scores,
    float* __restrict__ roi_labels)
{
  int bb = blockIdx.x, tid = threadIdx.x;
  __shared__ float bx[KTOP], by[KTOP];
  __shared__ int   kp[KTOP];
  for (int i = tid; i < KTOP; i += 512) {
    bx[i] = boxes[((long)bb * KTOP + i) * 7 + 0];
    by[i] = boxes[((long)bb * KTOP + i) * 7 + 1];
    kp[i] = keep[(long)bb * KTOP + i];
  }
  __syncthreads();
  for (int i = 0; i < KTOP; ++i) {
    if (kp[i]) {
      float xi = bx[i], yi = by[i];
      for (int j = i + 1 + tid; j < KTOP; j += 512) {
        if (kp[j]) {
          float dx = xi - bx[j], dy = yi - by[j];
          if (dx * dx + dy * dy < 4.0f) kp[j] = 0;
        }
      }
    }
    __syncthreads();
  }
  for (int k2 = tid; k2 < KTOP; k2 += 512) {
    long kk = (long)bb * KTOP + k2;
    float kf = kp[k2] ? 1.f : 0.f;
    for (int d = 0; d < 7; ++d) rois[kk * 7 + d] = boxes[kk * 7 + d] * kf;
    roi_scores[kk] = fsc[kk] * kf;
    roi_labels[kk] = kp[k2] ? (float)fcls[kk] : 0.f;
  }
}

// ---------------------------------------------------------------------------
extern "C" void kernel_launch(void* const* d_in, const int* in_sizes, int n_in,
                              void* d_out, int out_size, void* d_ws,
                              size_t ws_size, hipStream_t stream) {
  (void)in_sizes; (void)n_in; (void)out_size; (void)ws_size;

  const float* x    = (const float*)d_in[0];
  const float* sh_w = (const float*)d_in[1];
  const float* sh_g = (const float*)d_in[2];
  const float* sh_b = (const float*)d_in[3];
  const float* sh_m = (const float*)d_in[4];
  const float* sh_v = (const float*)d_in[5];
  static const int COUT[5] = {2, 1, 3, 2, 3};

  float* out = (float*)d_out;
  float* out_center = out;                         // [4,2,256,256]
  float* out_cz     = out + (long)NB * 2 * HWSZ;   // [4,1,256,256]
  float* out_dim    = out_cz + (long)NB * 1 * HWSZ;
  float* out_rot    = out_dim + (long)NB * 3 * HWSZ;
  float* out_hm     = out_rot + (long)NB * 2 * HWSZ;
  float* out_rois   = out_hm + (long)NB * 3 * HWSZ;
  float* out_rsc    = out_rois + (long)NB * KTOP * 7;
  float* out_rlb    = out_rsc + (long)NB * KTOP;
  float* branch_out[5] = {out_center, out_cz, out_dim, out_rot, out_hm};

  // ---- workspace carve-up ----
  char* ws = (char*)d_ws;
  size_t o = 0;
  auto carve = [&](size_t bytes) -> void* {
    void* p = ws + o;
    o = (o + bytes + 255) & ~(size_t)255;
    return p;
  };
  bf16_t* feat   = (bf16_t*)carve((size_t)NB * C_SH * HWSZ * 2);
  bf16_t* hidden = (bf16_t*)carve((size_t)NB * C_SH * HWSZ * 2);
  bf16_t* wsh    = (bf16_t*)carve((size_t)C_SH * C_IN * 9 * 2);
  bf16_t* wb1    = (bf16_t*)carve((size_t)C_SH * C_SH * 9 * 2);
  float*  scores = (float*)carve((size_t)NB * NCLS * HWSZ * 4);
  int*    taken  = (int*)carve((size_t)NB * NCLS * HWSZ * 4);
  float*  cls_s  = (float*)carve((size_t)NB * NCLS * KTOP * 4);
  int*    cls_i  = (int*)carve((size_t)NB * NCLS * KTOP * 4);
  float*  fsc    = (float*)carve((size_t)NB * KTOP * 4);
  int*    find   = (int*)carve((size_t)NB * KTOP * 4);
  int*    fcls   = (int*)carve((size_t)NB * KTOP * 4);
  float*  boxes  = (float*)carve((size_t)NB * KTOP * 7 * 4);
  int*    keep   = (int*)carve((size_t)NB * KTOP * 4);

  const int conv_grid = NB * (HT / 2) * (WT / 32);  // 4096 blocks

  // 1) shared conv: fp32 x (384ch) -> bf16 feat (64ch), BN+ReLU
  {
    int nw = C_SH * C_IN * 9;
    reorder_w_kernel<<<(nw + 255) / 256, 256, 0, stream>>>(sh_w, wsh, C_IN);
    conv3x3_bn_relu_wmma<C_IN, true><<<conv_grid, 256, 0, stream>>>(
        x, nullptr, wsh, sh_g, sh_b, sh_m, sh_v, feat);
  }

  // 2) five branches: 64->64 WMMA conv + BN+ReLU, then tiny head conv
  for (int br = 0; br < 5; ++br) {
    int base = 6 + br * 7;
    const float* w1 = (const float*)d_in[base + 0];
    const float* g  = (const float*)d_in[base + 1];
    const float* be = (const float*)d_in[base + 2];
    const float* me = (const float*)d_in[base + 3];
    const float* va = (const float*)d_in[base + 4];
    const float* w2 = (const float*)d_in[base + 5];
    const float* b2 = (const float*)d_in[base + 6];

    int nw = C_SH * C_SH * 9;
    reorder_w_kernel<<<(nw + 255) / 256, 256, 0, stream>>>(w1, wb1, C_SH);
    conv3x3_bn_relu_wmma<C_SH, false><<<conv_grid, 256, 0, stream>>>(
        nullptr, feat, wb1, g, be, me, va, hidden);
    conv3x3_head_kernel<<<((long)NB * HWSZ + 255) / 256, 256, 0, stream>>>(
        hidden, w2, b2, branch_out[br], COUT[br]);
  }

  // 3) decode: sigmoid, per-class topK, merge, boxes, NMS, emit
  {
    int n = NB * NCLS * HWSZ;
    sigmoid_kernel<<<(n + 255) / 256, 256, 0, stream>>>(out_hm, scores, n);
    topk_class_kernel<<<NB * NCLS, 256, 0, stream>>>(scores, taken, cls_s, cls_i);
    topk_merge_kernel<<<NB, 256, 0, stream>>>(cls_s, cls_i, fsc, find, fcls);
    decode_kernel<<<NB, 512, 0, stream>>>(out, fsc, find, fcls, boxes, keep);
    nms_emit_kernel<<<NB, 512, 0, stream>>>(boxes, fsc, fcls, keep,
                                            out_rois, out_rsc, out_rlb);
  }
}
